// Spatial_Binder_21646635172283
// MI455X (gfx1250) — matrix operand order repaired and checked by reading, not compiled, not run on hardware
//
#include <hip/hip_runtime.h>
#include <hip/hip_bf16.h>

typedef _Float16 half_t;
typedef __attribute__((ext_vector_type(16))) _Float16 v16h;
typedef __attribute__((ext_vector_type(8)))  float    v8f;
typedef __attribute__((ext_vector_type(4)))  float    f4_t;
typedef __attribute__((ext_vector_type(4)))  int      i4_t;

#define NSLOT   8
#define DSLOT   128
#define DIN     768
#define NTOK    441
#define NBATCH  40
#define SITERS  3
#define NROWS   (NBATCH * NTOK)        // 17640
#define NBS     (NBATCH * NSLOT)       // 320

__device__ __constant__ const float kSIGMA = 5.0f;
__device__ __constant__ const float kSCALE = 0.08838834764831845f; // 128^-0.5
__device__ __constant__ const float kEPS   = 1e-8f;

// CDNA5 async global->LDS copy (ASYNCcnt-tracked), if the toolchain declares it.
#if defined(__has_builtin)
#  if __has_builtin(__builtin_amdgcn_global_load_async_to_lds_b128)
#    define USE_ASYNC_LDS 1
#  endif
#endif
#ifndef USE_ASYNC_LDS
#  define USE_ASYNC_LDS 0
#endif

// ---------------------------------------------------------------------------
// WMMA fragment loaders (CDNA5 ISA 7.12.2 layouts, wave32)
// A 16x32 f16: lanes0-15 row=lane, K = {0..7,16..23}; lanes16-31 K={8..15,24..31}
// B 32x16 f16: lanes0-15 col=lane, K = {0..7,16..23}; lanes16-31 K={8..15,24..31}
// ---------------------------------------------------------------------------
__device__ inline v16h load_a_frag(const half_t* base, int ldh, int lane) {
  union { v16h v; unsigned u[8]; } f;
  int r  = lane & 15;
  int kb = (lane >> 4) << 3;           // 0 or 8
  const half_t* rp = base + r * ldh;
#pragma unroll
  for (int j = 0; j < 8; ++j) {
    int K = ((j >> 2) << 4) + kb + ((j & 3) << 1);
    f.u[j] = *(const unsigned*)(rp + K);
  }
  return f.v;
}

__device__ inline v16h load_b_frag(const half_t* base, int ldh, int lane) {
  union { v16h v; unsigned u[8]; } f;
  int c  = lane & 15;
  int kb = (lane >> 4) << 3;           // 0 or 8
  const half_t* cp = base + c * ldh + kb;
#pragma unroll
  for (int j = 0; j < 8; ++j)
    f.u[j] = *(const unsigned*)(cp + ((j >> 2) << 4) + ((j & 3) << 1));
  return f.v;
}

// ---------------------------------------------------------------------------
// Tiled WMMA GEMM: C[M,N] = act(A[M,K] @ B[K,N] + bias), A/B f16, acc fp32.
// Block tile 128x128x32, 256 threads (8 waves), wave tile 32x64 (2x4 WMMA).
// Double-buffered LDS; A-tile copy uses async-to-LDS when available.
// N, K must be multiples of 128/32 (true for all uses); M guarded (clamped
// loads, guarded stores).
// ---------------------------------------------------------------------------
template<bool RELU, bool OUT_HALF>
__global__ __launch_bounds__(256) void gemm_wmma_kernel(
    const half_t* __restrict__ A, const half_t* __restrict__ Bm,
    const float* __restrict__ bias, void* __restrict__ Cout,
    int M, int N, int K)
{
  constexpr int BM = 128, BN = 128, BK = 32, LDH = BK + 8;
  __shared__ half_t sA[2][BM * LDH];
  __shared__ half_t sB[2][BN * LDH];

  const int tid  = threadIdx.x;
  const int wid  = tid >> 5;
  const int lane = tid & 31;
  const int m0   = blockIdx.y * BM;
  const int n0   = blockIdx.x * BN;
  const int wr   = (wid >> 1) * 32;    // wave row offset in block tile
  const int wc   = (wid & 1) * 64;     // wave col offset in block tile

  v8f acc[2][4];
  for (int i = 0; i < 2; ++i)
    for (int j = 0; j < 4; ++j)
      for (int q = 0; q < 8; ++q) acc[i][j][q] = 0.f;

  // A tile copy: 128x32 halfs = 512 chunks of 8 halfs (16B), branch-free via clamp.
  auto load_a_tile = [&](int k0, int buf) {
    for (int c = tid; c < 512; c += 256) {
      int row = c >> 2, col = (c & 3) << 3;
      int gm = m0 + row;
      if (gm >= M) gm = M - 1;                    // clamp: no exec-mask juggling
      const half_t* src = A + (size_t)gm * K + k0 + col;
      half_t* dst = &sA[buf][row * LDH + col];
#if USE_ASYNC_LDS
      __builtin_amdgcn_global_load_async_to_lds_b128(
          (__attribute__((address_space(1))) i4_t*)(void*)const_cast<half_t*>(src),
          (__attribute__((address_space(3))) i4_t*)(void*)dst,
          0, 0);
#else
      union { f4_t f4; unsigned u[4]; } p;
      p.f4 = *(const f4_t*)src;
      *(unsigned*)(dst + 0) = p.u[0]; *(unsigned*)(dst + 2) = p.u[1];
      *(unsigned*)(dst + 4) = p.u[2]; *(unsigned*)(dst + 6) = p.u[3];
#endif
    }
  };
  // B tile: 32x128 halfs, transposed into LDS K-major per column.
  auto load_b_tile = [&](int k0, int buf) {
    for (int c = tid; c < 512; c += 256) {
      int krow = c >> 4, ncol = (c & 15) << 3;
      union { f4_t f4; half_t h[8]; } p;
      p.f4 = *(const f4_t*)(Bm + (size_t)(k0 + krow) * N + n0 + ncol);
#pragma unroll
      for (int j = 0; j < 8; ++j) sB[buf][(ncol + j) * LDH + krow] = p.h[j];
    }
  };
  auto stage_sync = [&]() {
#if USE_ASYNC_LDS
    asm volatile("s_wait_asynccnt 0x0" ::: "memory");
#endif
    __syncthreads();
  };

  int cur = 0;
  load_a_tile(0, cur);
  load_b_tile(0, cur);
  stage_sync();

  for (int k0 = 0; k0 < K; k0 += BK) {
    int nxt = k0 + BK;
    if (nxt < K) {                      // prefetch next tile into other buffer
      load_a_tile(nxt, cur ^ 1);
      load_b_tile(nxt, cur ^ 1);
    }
    v16h af0 = load_a_frag(&sA[cur][wr * LDH], LDH, lane);
    v16h af1 = load_a_frag(&sA[cur][(wr + 16) * LDH], LDH, lane);
#pragma unroll
    for (int ct = 0; ct < 4; ++ct) {
      v16h bf = load_b_frag(&sB[cur][(wc + ct * 16) * LDH], LDH, lane);
      acc[0][ct] = __builtin_amdgcn_wmma_f32_16x16x32_f16(
          false, af0, false, bf, (short)0, acc[0][ct], false, false);
      acc[1][ct] = __builtin_amdgcn_wmma_f32_16x16x32_f16(
          false, af1, false, bf, (short)0, acc[1][ct], false, false);
    }
    if (nxt < K) {
      stage_sync();
      cur ^= 1;
    }
  }

  // Epilogue: D layout lane L -> col=L&15; VGPR j -> row = j + 8*(L>=16)
#pragma unroll
  for (int rt = 0; rt < 2; ++rt) {
#pragma unroll
    for (int ct = 0; ct < 4; ++ct) {
      int col   = n0 + wc + ct * 16 + (lane & 15);
      int rbase = m0 + wr + rt * 16 + ((lane >> 4) << 3);
#pragma unroll
      for (int j = 0; j < 8; ++j) {
        int row = rbase + j;
        if (row < M) {
          float v = acc[rt][ct][j];
          if (bias) v += bias[col];
          if (RELU) v = fmaxf(v, 0.f);
          if (OUT_HALF) ((half_t*)Cout)[(size_t)row * N + col] = (half_t)v;
          else          ((float*) Cout)[(size_t)row * N + col] = v;
        }
      }
    }
  }
}

// ---------------------------------------------------------------------------
// LayerNorm rows -> f16 (one block per row)
// ---------------------------------------------------------------------------
__global__ void ln_rows_to_half_kernel(const float* __restrict__ x,
                                       const float* __restrict__ g,
                                       const float* __restrict__ b,
                                       half_t* __restrict__ out, int D)
{
  __shared__ float red[256];
  __shared__ float s_m, s_r;
  int row = blockIdx.x, tid = threadIdx.x, nt = blockDim.x;
  const float* xr = x + (size_t)row * D;
  float s = 0.f;
  for (int i = tid; i < D; i += nt) s += xr[i];
  red[tid] = s; __syncthreads();
  for (int o = nt >> 1; o > 0; o >>= 1) { if (tid < o) red[tid] += red[tid + o]; __syncthreads(); }
  if (tid == 0) s_m = red[0] / (float)D;
  __syncthreads();
  float m = s_m;
  s = 0.f;
  for (int i = tid; i < D; i += nt) { float d = xr[i] - m; s += d * d; }
  red[tid] = s; __syncthreads();
  for (int o = nt >> 1; o > 0; o >>= 1) { if (tid < o) red[tid] += red[tid + o]; __syncthreads(); }
  if (tid == 0) s_r = rsqrtf(red[0] / (float)D + 1e-5f);
  __syncthreads();
  float rstd = s_r;
  half_t* orow = out + (size_t)row * D;
  for (int i = tid; i < D; i += nt)
    orow[i] = (half_t)((xr[i] - m) * rstd * g[i] + b[i]);
}

__global__ void f32_to_f16_kernel(const float* __restrict__ in,
                                  half_t* __restrict__ out, int n)
{
  for (int i = blockIdx.x * blockDim.x + threadIdx.x; i < n; i += gridDim.x * blockDim.x)
    out[i] = (half_t)in[i];
}

// C[128,128] = A[128,128] @ B[128,128]  (tiny fp32)
__global__ void matmul128_kernel(const float* __restrict__ A,
                                 const float* __restrict__ B,
                                 float* __restrict__ C)
{
  int j = blockIdx.x * 16 + threadIdx.x;
  int i = blockIdx.y * 16 + threadIdx.y;
  float s = 0.f;
  for (int e = 0; e < 128; ++e) s += A[i * 128 + e] * B[e * 128 + j];
  C[i * 128 + j] = s;
}

// G1[2,128] = Wg @ Wf1 ; c1[128] = bg @ Wf1 + bf1
__global__ void g1c1_kernel(const float* __restrict__ Wg, const float* __restrict__ bg,
                            const float* __restrict__ Wf1, const float* __restrict__ bf1,
                            float* __restrict__ G1, float* __restrict__ c1)
{
  int j = threadIdx.x;
  float a0 = 0.f, a1 = 0.f, cc = 0.f;
  for (int e = 0; e < 128; ++e) {
    float w = Wf1[e * 128 + j];
    a0 += Wg[e] * w;
    a1 += Wg[128 + e] * w;
    cc += bg[e] * w;
  }
  G1[j] = a0; G1[128 + j] = a1; c1[j] = cc + bf1[j];
}

__global__ void gather_ag_kernel(const int* __restrict__ tok,
                                 const float* __restrict__ grid,
                                 float* __restrict__ ag, int n)
{
  int i = blockIdx.x * blockDim.x + threadIdx.x;
  if (i < n) {
    int idx = tok[i];
    ag[2 * i + 0] = grid[2 * idx + 0];
    ag[2 * i + 1] = grid[2 * idx + 1];
  }
}

__global__ void init_state_kernel(const float* __restrict__ s0,
                                  const float* __restrict__ p0,
                                  const float* __restrict__ ss0,
                                  float* __restrict__ slots,
                                  float* __restrict__ S_p,
                                  float* __restrict__ S_s)
{
  int i = blockIdx.x * blockDim.x + threadIdx.x;
  if (i < NBS * DSLOT) slots[i] = s0[i & (NSLOT * DSLOT - 1)];
  if (i < NBS * 2) { S_p[i] = p0[i & 15]; S_s[i] = ss0[i & 15]; }
}

// per (b,s): sn = LN(slot); q = sn@WQ; qW2[e] = sum_d Wf2[e,d] q[d]; qb2 = q.bf2
__global__ __launch_bounds__(128) void q_proj_kernel(
    const float* __restrict__ slots, const float* __restrict__ ln_g,
    const float* __restrict__ ln_b, const float* __restrict__ WQ,
    const float* __restrict__ Wf2, const float* __restrict__ bf2,
    float* __restrict__ qW2, float* __restrict__ qb2)
{
  __shared__ float sn[128], qv[128], red[128];
  int rs = blockIdx.x, d = threadIdx.x;
  float x = slots[rs * 128 + d];
  red[d] = x; __syncthreads();
  for (int o = 64; o > 0; o >>= 1) { if (d < o) red[d] += red[d + o]; __syncthreads(); }
  float m = red[0] * (1.f / 128.f); __syncthreads();
  float xm = x - m;
  red[d] = xm * xm; __syncthreads();
  for (int o = 64; o > 0; o >>= 1) { if (d < o) red[d] += red[d + o]; __syncthreads(); }
  float rstd = rsqrtf(red[0] * (1.f / 128.f) + 1e-5f); __syncthreads();
  sn[d] = xm * rstd * ln_g[d] + ln_b[d]; __syncthreads();
  float q = 0.f;
  for (int e = 0; e < 128; ++e) q += sn[e] * WQ[e * 128 + d];
  qv[d] = q; __syncthreads();
  float w = 0.f;
  for (int e = 0; e < 128; ++e) w += Wf2[d * 128 + e] * qv[e];
  qW2[rs * 128 + d] = w;
  red[d] = qv[d] * bf2[d]; __syncthreads();
  for (int o = 64; o > 0; o >>= 1) { if (d < o) red[d] += red[d + o]; __syncthreads(); }
  if (d == 0) qb2[rs] = red[0];
}

// dots[b,s,n] = SCALE*( relu(A_k + rel@G1 + c1) . qW2[b,s] + qb2[b,s] )
__global__ __launch_bounds__(256) void attn_dots_kernel(
    const float* __restrict__ Ak, const float* __restrict__ G1,
    const float* __restrict__ c1, const float* __restrict__ qW2,
    const float* __restrict__ qb2, const float* __restrict__ ag,
    const float* __restrict__ S_p, const float* __restrict__ S_s,
    float* __restrict__ attn)
{
  int item = blockIdx.x * 8 + (threadIdx.x >> 5);
  int lane = threadIdx.x & 31;
  if (item >= NBATCH * NSLOT * NTOK) return;
  int b = item / (NSLOT * NTOK);
  int r = item - b * NSLOT * NTOK;
  int s = r / NTOK;
  int n = r - s * NTOK;
  int bs = b * NSLOT + s;
  int bn = b * NTOK + n;
  float agx = ag[2 * bn], agy = ag[2 * bn + 1];
  float r0 = (agx - S_p[bs * 2 + 0]) / (S_s[bs * 2 + 0] * kSIGMA);
  float r1 = (agy - S_p[bs * 2 + 1]) / (S_s[bs * 2 + 1] * kSIGMA);
  int e = lane * 4;
  f4_t a4 = *(const f4_t*)(Ak + (size_t)bn * 128 + e);
  f4_t g0 = *(const f4_t*)(G1 + e);
  f4_t g1 = *(const f4_t*)(G1 + 128 + e);
  f4_t cc = *(const f4_t*)(c1 + e);
  f4_t qw = *(const f4_t*)(qW2 + (size_t)bs * 128 + e);
  float sum = 0.f;
#pragma unroll
  for (int j = 0; j < 4; ++j) {
    float h = a4[j] + r0 * g0[j] + r1 * g1[j] + cc[j];
    sum += fmaxf(h, 0.f) * qw[j];
  }
  for (int o = 16; o > 0; o >>= 1) sum += __shfl_xor(sum, o, 32);
  if (lane == 0)
    attn[(size_t)bs * NTOK + n] = kSCALE * (sum + qb2[bs]);
}

// softmax over slots (8) for each (b,n); + EPS
__global__ void softmax_slots_kernel(float* __restrict__ attn)
{
  int i = blockIdx.x * blockDim.x + threadIdx.x;
  if (i >= NBATCH * NTOK) return;
  int b = i / NTOK, n = i - b * NTOK;
  float v[NSLOT], mx = -3.4e38f;
#pragma unroll
  for (int s = 0; s < NSLOT; ++s) {
    v[s] = attn[((size_t)(b * NSLOT + s)) * NTOK + n];
    mx = fmaxf(mx, v[s]);
  }
  float sum = 0.f;
#pragma unroll
  for (int s = 0; s < NSLOT; ++s) { v[s] = expf(v[s] - mx); sum += v[s]; }
  float inv = 1.f / sum;
#pragma unroll
  for (int s = 0; s < NSLOT; ++s)
    attn[((size_t)(b * NSLOT + s)) * NTOK + n] = v[s] * inv + kEPS;
}

// renormalize each (b,s) row over tokens
__global__ __launch_bounds__(256) void renorm_kernel(float* __restrict__ attn)
{
  __shared__ float red[256];
  __shared__ float s_inv;
  int bs = blockIdx.x, tid = threadIdx.x;
  float* row = attn + (size_t)bs * NTOK;
  float s = 0.f;
  for (int i = tid; i < NTOK; i += 256) s += row[i];
  red[tid] = s; __syncthreads();
  for (int o = 128; o > 0; o >>= 1) { if (tid < o) red[tid] += red[tid + o]; __syncthreads(); }
  if (tid == 0) s_inv = 1.f / red[0];
  __syncthreads();
  float inv = s_inv;
  for (int i = tid; i < NTOK; i += 256) row[i] *= inv;
}

// uh[bs,e] = sum_n attn * relu(A_v + rel@G1 + c1)
__global__ __launch_bounds__(128) void agg_kernel(
    const float* __restrict__ Av, const float* __restrict__ G1,
    const float* __restrict__ c1, const float* __restrict__ ag,
    const float* __restrict__ S_p, const float* __restrict__ S_s,
    const float* __restrict__ attn, float* __restrict__ uh)
{
  int bs = blockIdx.x, b = bs >> 3, e = threadIdx.x;
  float px = S_p[bs * 2 + 0], py = S_p[bs * 2 + 1];
  float isx = 1.f / (S_s[bs * 2 + 0] * kSIGMA);
  float isy = 1.f / (S_s[bs * 2 + 1] * kSIGMA);
  float g0 = G1[e], g1 = G1[128 + e], cc = c1[e];
  float acc = 0.f;
  for (int n = 0; n < NTOK; ++n) {
    int bn = b * NTOK + n;
    float a = attn[(size_t)bs * NTOK + n];
    float r0 = (ag[2 * bn] - px) * isx;
    float r1 = (ag[2 * bn + 1] - py) * isy;
    float h = Av[(size_t)bn * 128 + e] + r0 * g0 + r1 * g1 + cc;
    acc += a * fmaxf(h, 0.f);
  }
  uh[(size_t)bs * 128 + e] = acc;
}

// S_p = sum attn*ag ; S_s = sqrt(sum attn*(ag - S_p)^2)
__global__ __launch_bounds__(256) void pos_kernel(
    const float* __restrict__ ag, const float* __restrict__ attn,
    float* __restrict__ S_p, float* __restrict__ S_s)
{
  __shared__ float red[256];
  __shared__ float res[5];
  int bs = blockIdx.x, b = bs >> 3, tid = threadIdx.x;
  float sa = 0.f, sx = 0.f, sy = 0.f, sxx = 0.f, syy = 0.f;
  for (int n = tid; n < NTOK; n += 256) {
    float a = attn[(size_t)bs * NTOK + n];
    float x = ag[2 * (b * NTOK + n)], y = ag[2 * (b * NTOK + n) + 1];
    sa += a; sx += a * x; sy += a * y; sxx += a * x * x; syy += a * y * y;
  }
  float vals[5] = {sa, sx, sy, sxx, syy};
  for (int v = 0; v < 5; ++v) {
    red[tid] = vals[v]; __syncthreads();
    for (int o = 128; o > 0; o >>= 1) { if (tid < o) red[tid] += red[tid + o]; __syncthreads(); }
    if (tid == 0) res[v] = red[0];
    __syncthreads();
  }
  if (tid == 0) {
    float SA = res[0], px = res[1], py = res[2], xx = res[3], yy = res[4];
    S_p[bs * 2 + 0] = px; S_p[bs * 2 + 1] = py;
    float vx = xx - 2.f * px * px + px * px * SA;
    float vy = yy - 2.f * py * py + py * py * SA;
    S_s[bs * 2 + 0] = sqrtf(fmaxf(vx, 0.f));
    S_s[bs * 2 + 1] = sqrtf(fmaxf(vy, 0.f));
  }
}

__device__ inline float sigmoidf_(float x) { return 1.f / (1.f + expf(-x)); }

// updates = uh@Wf2+bf2 ; GRUCell ; slots += MLP(LN(slots))
__global__ __launch_bounds__(128) void gru_mlp_kernel(
    const float* __restrict__ uh, const float* __restrict__ Wf2,
    const float* __restrict__ bf2, const float* __restrict__ w_ih,
    const float* __restrict__ w_hh, const float* __restrict__ b_ih,
    const float* __restrict__ b_hh, const float* __restrict__ mg,
    const float* __restrict__ mb, const float* __restrict__ w1,
    const float* __restrict__ b1, const float* __restrict__ w2,
    const float* __restrict__ b2, float* __restrict__ slots)
{
  __shared__ float su[128], upd[128], h[128], gi[384], gh[384];
  __shared__ float hn[128], y[128], hid[512], red[128];
  int bs = blockIdx.x, d = threadIdx.x;
  su[d] = uh[(size_t)bs * 128 + d];
  h[d]  = slots[(size_t)bs * 128 + d];
  __syncthreads();
  float t = 0.f;
  for (int e = 0; e < 128; ++e) t += su[e] * Wf2[e * 128 + d];
  upd[d] = t + bf2[d];
  __syncthreads();
  for (int j = d; j < 384; j += 128) {
    float a = 0.f, c = 0.f;
    for (int e = 0; e < 128; ++e) {
      a += upd[e] * w_ih[e * 384 + j];
      c += h[e]   * w_hh[e * 384 + j];
    }
    gi[j] = a + b_ih[j];
    gh[j] = c + b_hh[j];
  }
  __syncthreads();
  float rr = sigmoidf_(gi[d] + gh[d]);
  float zz = sigmoidf_(gi[128 + d] + gh[128 + d]);
  float nn = tanhf(gi[256 + d] + rr * gh[256 + d]);
  float hnew = (1.f - zz) * nn + zz * h[d];
  hn[d] = hnew;
  red[d] = hnew; __syncthreads();
  for (int o = 64; o > 0; o >>= 1) { if (d < o) red[d] += red[d + o]; __syncthreads(); }
  float m = red[0] * (1.f / 128.f); __syncthreads();
  float xm = hnew - m;
  red[d] = xm * xm; __syncthreads();
  for (int o = 64; o > 0; o >>= 1) { if (d < o) red[d] += red[d + o]; __syncthreads(); }
  float rstd = rsqrtf(red[0] * (1.f / 128.f) + 1e-5f); __syncthreads();
  y[d] = xm * rstd * mg[d] + mb[d];
  __syncthreads();
  for (int j = d; j < 512; j += 128) {
    float a = 0.f;
    for (int e = 0; e < 128; ++e) a += y[e] * w1[e * 512 + j];
    hid[j] = fmaxf(a + b1[j], 0.f);
  }
  __syncthreads();
  float o = 0.f;
  for (int j = 0; j < 512; ++j) o += hid[j] * w2[j * 128 + d];
  slots[(size_t)bs * 128 + d] = hn[d] + o + b2[d];
}

__global__ __launch_bounds__(128) void final_slots_kernel(
    const float* __restrict__ slots, const float* __restrict__ fw,
    const float* __restrict__ fb, float* __restrict__ out)
{
  __shared__ float sr[128];
  int bs = blockIdx.x, d = threadIdx.x;
  sr[d] = slots[(size_t)bs * 128 + d];
  __syncthreads();
  float o = 0.f;
  for (int e = 0; e < 128; ++e) o += sr[e] * fw[e * 128 + d];
  out[(size_t)bs * 128 + d] = o + fb[d];
}

__global__ void attn_out_kernel(const float* __restrict__ attn, float* __restrict__ out)
{
  int i = blockIdx.x * blockDim.x + threadIdx.x;
  if (i >= 8 * NSLOT * NTOK) return;
  int c = i / (NSLOT * NTOK);
  int rem = i - c * NSLOT * NTOK;
  int s = rem / NTOK;
  int n = rem - s * NTOK;
  int b = c * 5 + 2;                       // center frame
  out[i] = attn[((size_t)(b * NSLOT + s)) * NTOK + n];
}

// ---------------------------------------------------------------------------
extern "C" void kernel_launch(void* const* d_in, const int* in_sizes, int n_in,
                              void* d_out, int out_size, void* d_ws, size_t ws_size,
                              hipStream_t stream)
{
  const float* inputs   = (const float*)d_in[0];
  const int*   tok      = (const int*)  d_in[1];
  const float* abs_grid = (const float*)d_in[2];
  const float* slots0   = (const float*)d_in[3];
  const float* Sp0      = (const float*)d_in[4];
  const float* Ss0      = (const float*)d_in[5];
  const float* WQ       = (const float*)d_in[6];
  const float* ln_g     = (const float*)d_in[7];
  const float* ln_b     = (const float*)d_in[8];
  const float* gru_wih  = (const float*)d_in[9];
  const float* gru_whh  = (const float*)d_in[10];
  const float* gru_bih  = (const float*)d_in[11];
  const float* gru_bhh  = (const float*)d_in[12];
  const float* mlp_lg   = (const float*)d_in[13];
  const float* mlp_lb   = (const float*)d_in[14];
  const float* mlp_w1   = (const float*)d_in[15];
  const float* mlp_b1   = (const float*)d_in[16];
  const float* mlp_w2   = (const float*)d_in[17];
  const float* mlp_b2   = (const float*)d_in[18];
  const float* WK       = (const float*)d_in[19];
  const float* WV       = (const float*)d_in[20];
  const float* Wg       = (const float*)d_in[21];
  const float* bg       = (const float*)d_in[22];
  const float* Wf1      = (const float*)d_in[23];
  const float* bf1      = (const float*)d_in[24];
  const float* Wf2      = (const float*)d_in[25];
  const float* bf2      = (const float*)d_in[26];
  const float* im_l1g   = (const float*)d_in[27];
  const float* im_l1b   = (const float*)d_in[28];
  const float* im_w1    = (const float*)d_in[29];
  const float* im_b1    = (const float*)d_in[30];
  const float* im_w2    = (const float*)d_in[31];
  const float* im_b2    = (const float*)d_in[32];
  const float* im_l2g   = (const float*)d_in[33];
  const float* im_l2b   = (const float*)d_in[34];
  const float* fin_w    = (const float*)d_in[35];
  const float* fin_b    = (const float*)d_in[36];
  float* out = (float*)d_out;

  // ---- workspace carve (256B aligned) ----
  char* base = (char*)d_ws;
  size_t off = 0;
  auto carve = [&](size_t bytes) -> void* {
    void* p = base + off;
    off += (bytes + 255) & ~(size_t)255;
    return p;
  };
  half_t* ln_x_h = (half_t*)carve((size_t)NROWS * DIN * 2);    // 27.1 MB (reused as A_k/A_v)
  half_t* x1_h   = (half_t*)carve((size_t)NROWS * DIN * 2);    // 27.1 MB
  half_t* w1_h   = (half_t*)carve((size_t)DIN * DIN * 2);
  half_t* w2_h   = (half_t*)carve((size_t)DIN * DSLOT * 2);
  float*  x2_f   = (float*) carve((size_t)NROWS * DSLOT * 4);
  half_t* x_h    = (half_t*)carve((size_t)NROWS * DSLOT * 2);
  float*  Mk_f   = (float*) carve(128 * 128 * 4);
  float*  Mv_f   = (float*) carve(128 * 128 * 4);
  half_t* Mk_h   = (half_t*)carve(128 * 128 * 2);
  half_t* Mv_h   = (half_t*)carve(128 * 128 * 2);
  float*  G1     = (float*) carve(2 * 128 * 4);
  float*  c1     = (float*) carve(128 * 4);
  float*  ag     = (float*) carve((size_t)NROWS * 2 * 4);
  float*  slots  = (float*) carve((size_t)NBS * DSLOT * 4);
  float*  S_p    = (float*) carve((size_t)NBS * 2 * 4);
  float*  S_s    = (float*) carve((size_t)NBS * 2 * 4);
  float*  qW2    = (float*) carve((size_t)NBS * DSLOT * 4);
  float*  qb2    = (float*) carve((size_t)NBS * 4);
  float*  attn   = (float*) carve((size_t)NBS * NTOK * 4);
  float*  uh     = (float*) carve((size_t)NBS * DSLOT * 4);
  // A_k / A_v alias the (dead-after-GEMM1) ln_x_h region: 2*9.03 MB <= 27.1 MB
  float* A_k = (float*)ln_x_h;
  float* A_v = A_k + (size_t)NROWS * DSLOT;

  // ---- once-per-call precompute ----
  ln_rows_to_half_kernel<<<NROWS, 256, 0, stream>>>(inputs, im_l1g, im_l1b, ln_x_h, DIN);
  f32_to_f16_kernel<<<1024, 256, 0, stream>>>(im_w1, w1_h, DIN * DIN);
  f32_to_f16_kernel<<<256, 256, 0, stream>>>(im_w2, w2_h, DIN * DSLOT);

  dim3 g1((DIN + 127) / 128, (NROWS + 127) / 128);
  gemm_wmma_kernel<true, true><<<g1, 256, 0, stream>>>(ln_x_h, w1_h, im_b1, x1_h,
                                                       NROWS, DIN, DIN);
  dim3 g2(1, (NROWS + 127) / 128);
  gemm_wmma_kernel<false, false><<<g2, 256, 0, stream>>>(x1_h, w2_h, im_b2, x2_f,
                                                         NROWS, DSLOT, DIN);
  ln_rows_to_half_kernel<<<NROWS, 128, 0, stream>>>(x2_f, im_l2g, im_l2b, x_h, DSLOT);

  matmul128_kernel<<<dim3(8, 8), dim3(16, 16), 0, stream>>>(WK, Wf1, Mk_f);
  matmul128_kernel<<<dim3(8, 8), dim3(16, 16), 0, stream>>>(WV, Wf1, Mv_f);
  f32_to_f16_kernel<<<64, 256, 0, stream>>>(Mk_f, Mk_h, 128 * 128);
  f32_to_f16_kernel<<<64, 256, 0, stream>>>(Mv_f, Mv_h, 128 * 128);
  g1c1_kernel<<<1, 128, 0, stream>>>(Wg, bg, Wf1, bf1, G1, c1);

  gemm_wmma_kernel<false, false><<<g2, 256, 0, stream>>>(x_h, Mk_h, nullptr, A_k,
                                                         NROWS, DSLOT, DSLOT);
  gemm_wmma_kernel<false, false><<<g2, 256, 0, stream>>>(x_h, Mv_h, nullptr, A_v,
                                                         NROWS, DSLOT, DSLOT);

  gather_ag_kernel<<<(NROWS + 255) / 256, 256, 0, stream>>>(tok, abs_grid, ag, NROWS);
  init_state_kernel<<<(NBS * DSLOT + 255) / 256, 256, 0, stream>>>(slots0, Sp0, Ss0,
                                                                   slots, S_p, S_s);

  // ---- slot attention iterations ----
  for (int t = 0; t <= SITERS; ++t) {
    q_proj_kernel<<<NBS, 128, 0, stream>>>(slots, ln_g, ln_b, WQ, Wf2, bf2, qW2, qb2);
    attn_dots_kernel<<<(NBATCH * NSLOT * NTOK) / 8, 256, 0, stream>>>(
        A_k, G1, c1, qW2, qb2, ag, S_p, S_s, attn);
    softmax_slots_kernel<<<(NBATCH * NTOK + 255) / 256, 256, 0, stream>>>(attn);
    renorm_kernel<<<NBS, 256, 0, stream>>>(attn);
    if (t < SITERS) {
      agg_kernel<<<NBS, 128, 0, stream>>>(A_v, G1, c1, ag, S_p, S_s, attn, uh);
      pos_kernel<<<NBS, 256, 0, stream>>>(ag, attn, S_p, S_s);
      gru_mlp_kernel<<<NBS, 128, 0, stream>>>(uh, Wf2, bf2, gru_wih, gru_whh,
                                              gru_bih, gru_bhh, mlp_lg, mlp_lb,
                                              mlp_w1, mlp_b1, mlp_w2, mlp_b2, slots);
    }
  }

  // ---- outputs: slots [40,8,128] then attn center-frame [8,8,441] ----
  final_slots_kernel<<<NBS, 128, 0, stream>>>(slots, fin_w, fin_b, out);
  attn_out_kernel<<<(8 * NSLOT * NTOK + 255) / 256, 256, 0, stream>>>(
      attn, out + (size_t)NBS * DSLOT);
}
